// GraphAttentionLayer_51213190037996
// MI455X (gfx1250) — matrix operand order repaired and checked by reading, not compiled, not run on hardware
//
#include <hip/hip_runtime.h>
#include <hip/hip_bf16.h>

// ---------------------------------------------------------------------------
// Fused GAT layer for MI455X (gfx1250, wave32, WMMA), v3.
//   Pre-pass: Wt[n][k] = bf16(W[k][n])  (256 KB, L2-resident, in d_ws)
//   Main:     h = input @ W via v_wmma_f32_16x16x32_bf16; fragments loaded
//             directly from global (WGP$ catches the 4x intra-WG A reuse),
//             barrier-free K loop; adj tile + row counts in LDS; fused
//             attention-factor + ELU epilogue.
// attention[n,m] = cnt[n]==0 ? 1/256 : (adj[n,m]>0 ? 1/cnt[n] : 0)  (exact:
// the softmax logit is row-constant, so `a` never affects the output).
// Roofline: 820 MB min traffic -> ~35 us at 23.3 TB/s; bf16 WMMA keeps the
// 52.4 GFLOP GEMM under the memory roof (f32 16x16x4 would be compute-bound).
// ---------------------------------------------------------------------------

typedef __bf16 v16bf __attribute__((ext_vector_type(16)));
typedef __bf16 v8bf  __attribute__((ext_vector_type(8)));
typedef float  v8f   __attribute__((ext_vector_type(8)));
typedef float  v4f   __attribute__((ext_vector_type(4)));
typedef int    v4i   __attribute__((ext_vector_type(4)));

#define NROWS 200000
#define KDIM  512
#define MOUT  256
#define BM    32

// ---- W pre-pack: transpose + bf16 convert (512x256 -> Wt[256][512]) --------
__global__ __launch_bounds__(256)
void prep_wt_kernel(const float* __restrict__ W, __bf16* __restrict__ Wt)
{
    const int idx = blockIdx.x * 256 + threadIdx.x;   // 0 .. 512*256-1
    const int k = idx >> 8;          // 0..511
    const int n = idx & 255;         // 0..255
    Wt[(size_t)n * KDIM + k] = (__bf16)W[(size_t)k * MOUT + n];
}

// ---- Fused GEMM + attention + ELU ------------------------------------------
__global__ __launch_bounds__(256)
void gat_fused_kernel(const float* __restrict__ input,
                      const int*   __restrict__ adj,
                      const __bf16* __restrict__ Wt,
                      float*       __restrict__ out)
{
    __shared__ int AdjS[BM][MOUT];    // 32 KB
    __shared__ int cnt[BM];

    const int tid    = threadIdx.x;   // 0..255 (8 wave32)
    const int wave   = tid >> 5;
    const int lane   = tid & 31;
    const int laneM  = lane & 15;
    const int laneHi = lane >> 4;     // 0 or 1
    const int mt     = wave >> 2;     // 0..1 : 16-row M sub-tile
    const int nq     = wave & 3;      // 0..3 : 64-col quadrant
    const int rowBase = blockIdx.x * BM;

    // ---- Phase 0: adj tile -> LDS, per-row positive counts -----------------
    if (tid < BM) cnt[tid] = 0;
    __syncthreads();
    {
        const int r  = tid >> 3;                // 8 threads per row
        const int c0 = (tid & 7) * 32;          // 32 cols each
        const int* aptr = adj + (size_t)(rowBase + r) * MOUT + c0;
        int local = 0;
#pragma unroll
        for (int c = 0; c < 32; c += 4) {
            v4i av = *(const v4i*)(aptr + c);
            *(v4i*)&AdjS[r][c0 + c] = av;
            local += (av.x > 0) + (av.y > 0) + (av.z > 0) + (av.w > 0);
        }
        atomicAdd(&cnt[r], local);
    }

    // ---- Barrier-free K loop ------------------------------------------------
    v8f acc[4];
#pragma unroll
    for (int i = 0; i < 4; ++i) acc[i] = (v8f)0.0f;

    // A: this lane's row of the input (f32), fragment slices cvt'd to bf16.
    const float* aptr = input + (size_t)(rowBase + mt * 16 + laneM) * KDIM;
    // B: pre-packed bf16 Wt row per output column; K = {0..7, 16..23} (+8 for hi).
    const __bf16* bbase = Wt + (size_t)(nq * 64 + laneM) * KDIM + laneHi * 8;

#pragma unroll 4
    for (int ks = 0; ks < KDIM; ks += 32) {
        // A fragment (16x32 bf16): lanes 0-15 K = ks+{0..7, 16..23}; +8 for hi.
        const float* ap = aptr + ks + laneHi * 8;
        v4f a0 = *(const v4f*)(ap + 0);
        v4f a1 = *(const v4f*)(ap + 4);
        v4f a2 = *(const v4f*)(ap + 16);
        v4f a3 = *(const v4f*)(ap + 20);
        v16bf afrag;
#pragma unroll
        for (int i = 0; i < 4; ++i) {
            afrag[i]      = (__bf16)a0[i];
            afrag[i + 4]  = (__bf16)a1[i];
            afrag[i + 8]  = (__bf16)a2[i];
            afrag[i + 12] = (__bf16)a3[i];
        }

        // B fragments (32x16 bf16): lane holds col, k = {0..7, 16..23} (+8 for hi).
        v16bf bfrag[4];
#pragma unroll
        for (int nt = 0; nt < 4; ++nt) {
            const __bf16* bp = bbase + (size_t)nt * 16 * KDIM + ks;
            v8bf b0 = *(const v8bf*)(bp + 0);
            v8bf b1 = *(const v8bf*)(bp + 16);
#pragma unroll
            for (int i = 0; i < 8; ++i) { bfrag[nt][i] = b0[i]; bfrag[nt][i + 8] = b1[i]; }
        }

        // Prefetch next A slice into the NEAR cache (WGP$ serves the 4x reuse).
        if (ks + 32 < KDIM)
            __builtin_prefetch(aptr + ks + 32 + laneHi * 8, 0, 3);

#pragma unroll
        for (int nt = 0; nt < 4; ++nt) {
            acc[nt] = __builtin_amdgcn_wmma_f32_16x16x32_bf16(
                /*neg_a=*/false, afrag, /*neg_b=*/false, bfrag[nt],
                /*c_mod=*/(short)0, acc[nt],
                /*reuse_a=*/false, /*reuse_b=*/false);
        }
    }

    __syncthreads();   // cnt / AdjS complete before epilogue reads

    // ---- Epilogue: attention factor + ELU, then store ----------------------
    // C layout: VGPR r -> M = r + 8*laneHi, N = laneM (within 16x16 tile).
#pragma unroll
    for (int nt = 0; nt < 4; ++nt) {
        const int col = nq * 64 + nt * 16 + laneM;
#pragma unroll
        for (int r = 0; r < 8; ++r) {
            const int row  = mt * 16 + laneHi * 8 + r;
            const int cn   = cnt[row];
            const int adjv = AdjS[row][col];
            const float f  = (cn == 0) ? (1.0f / MOUT)
                                       : (adjv > 0 ? 1.0f / (float)cn : 0.0f);
            const float x  = f * acc[nt][r];
            out[(size_t)(rowBase + row) * MOUT + col] =
                (x > 0.0f) ? x : (__expf(x) - 1.0f);
        }
    }
}

extern "C" void kernel_launch(void* const* d_in, const int* in_sizes, int n_in,
                              void* d_out, int out_size, void* d_ws, size_t ws_size,
                              hipStream_t stream)
{
    const float* input = (const float*)d_in[0];   // [200000, 512] f32
    const int*   adj   = (const int*)  d_in[1];   // [200000, 256] i32
    const float* W     = (const float*)d_in[2];   // [512, 256]    f32
    // d_in[3] = a [512,1] f32 — provably does not affect the output.
    float* out = (float*)d_out;                   // [200000, 256] f32

    __bf16* Wt = (__bf16*)d_ws;                   // needs 512*256*2 = 256 KB

    // Pre-pack W every call (deterministic; ~0.5 MB read, negligible).
    prep_wt_kernel<<<dim3((KDIM * MOUT) / 256), dim3(256), 0, stream>>>(W, Wt);

    gat_fused_kernel<<<dim3(NROWS / BM), dim3(256), 0, stream>>>(input, adj, Wt, out);
}
